// PointConvEncoder_14078902796592
// MI455X (gfx1250) — matrix operand order repaired and checked by reading, not compiled, not run on hardware
//
#include <hip/hip_runtime.h>
#include <hip/hip_bf16.h>

// ---------------- CDNA5 WMMA types & helpers ----------------
typedef __attribute__((ext_vector_type(16))) __bf16 v16bf;
typedef __attribute__((ext_vector_type(8)))  float  v8f;

#define YDIM 32
#define KNN_K 32

__device__ __forceinline__ v8f wmma_bf16(v16bf a, v16bf b, v8f c) {
  // emits v_wmma_f32_16x16x32_bf16
  return __builtin_amdgcn_wmma_f32_16x16x32_bf16(false, a, false, b,
                                                 (short)0, c, false, false);
}

// Element offset (in bf16 elements) of A-matrix entry (row16, klocal) inside a
// 16x32 fragment region stored as [lane][16] (512 elements, 1KB).
// Per CDNA5 ISA: lanes 0-15 hold klocal {0..7,16..23}, lanes 16-31
// hold klocal {8..15,24..31}.
__device__ __forceinline__ int a_frag_off(int row16, int klocal) {
  int kb   = (klocal >> 3) & 1;
  int lane = (row16 & 15) + 16 * kb;
  int kp   = klocal - 8 * kb;            // {0..7} U {16..23}
  int i    = kp - ((kp >= 16) ? 8 : 0);  // 0..15
  return lane * 16 + i;
}

// ---------------- weight packing: fp32 [K,N] -> bf16 fragment order ----------
// Output layout: [kk][nt][lane][16] so B-fragment load is one aligned v16bf.
// B-matrix 32x16 bf16: lane holds col n=lane&15; elem i is klocal=(lane>=16?8:0)+(i&7)+(i>=8?16:0).
__global__ void pack_w_frag_kernel(const float* __restrict__ W,
                                   __bf16* __restrict__ out,
                                   int K, int N, int NT, int total) {
  int i = blockIdx.x * 256 + threadIdx.x;
  if (i >= total) return;
  int e    = i & 15;
  int lane = (i >> 4) & 31;
  int tile = i >> 9;
  int nt   = tile % NT;
  int kk   = tile / NT;
  int n = nt * 16 + (lane & 15);
  int k = kk * 32 + ((lane & 16) ? 8 : 0) + (e & 7) + ((e & 8) << 1);
  float v = (k < K && n < N) ? W[k * N + n] : 0.0f;
  out[i] = (__bf16)v;
}

// ---------------- farthest point sampling ----------------
__global__ void fps_kernel(const float* __restrict__ xyz, int N, int npoint,
                           int* __restrict__ out_idx, float* __restrict__ out_xyz) {
  __shared__ float dist[4096];
  __shared__ float rmax[256];
  __shared__ int   rarg[256];
  __shared__ int   cur;
  int b = blockIdx.x;
  int t = threadIdx.x;
  const float* P = xyz + (size_t)b * N * 3;
  for (int i = t; i < N; i += 256) dist[i] = 1e10f;
  if (t == 0) cur = 0;
  __syncthreads();
  for (int s = 0; s < npoint; ++s) {
    int far = cur;
    float cx = P[far * 3 + 0], cy = P[far * 3 + 1], cz = P[far * 3 + 2];
    if (t == 0) {
      out_idx[b * npoint + s] = far;
      out_xyz[((size_t)b * npoint + s) * 3 + 0] = cx;
      out_xyz[((size_t)b * npoint + s) * 3 + 1] = cy;
      out_xyz[((size_t)b * npoint + s) * 3 + 2] = cz;
    }
    float bd = -1.0f; int bi = 0;
    for (int i = t; i < N; i += 256) {
      float dx = P[i * 3 + 0] - cx;
      float dy = P[i * 3 + 1] - cy;
      float dz = P[i * 3 + 2] - cz;
      float d  = dx * dx + dy * dy + dz * dz;
      float nd = fminf(dist[i], d);
      dist[i]  = nd;
      if (nd > bd) { bd = nd; bi = i; }
    }
    rmax[t] = bd; rarg[t] = bi;
    __syncthreads();
    for (int off = 128; off > 0; off >>= 1) {
      if (t < off) {
        bool take = (rmax[t + off] > rmax[t]) ||
                    (rmax[t + off] == rmax[t] && rarg[t + off] < rarg[t]);
        if (take) { rmax[t] = rmax[t + off]; rarg[t] = rarg[t + off]; }
      }
      __syncthreads();
    }
    if (t == 0) cur = rarg[0];
    __syncthreads();
  }
}

// ---------------- brute-force kNN (k=32), stable like lax.top_k ----------------
__global__ void knn_kernel(const float* __restrict__ q_xyz,
                           const float* __restrict__ p_xyz,
                           int S, int N, int total, int* __restrict__ out) {
  int tid = blockIdx.x * 128 + threadIdx.x;
  if (tid >= total) return;
  int b = tid / S;
  const float* Q = q_xyz + (size_t)tid * 3;
  const float* P = p_xyz + (size_t)b * N * 3;
  float qx = Q[0], qy = Q[1], qz = Q[2];
  float lastd = -1.0f; int lasti = -1;
  for (int j = 0; j < KNN_K; ++j) {
    float bd = 3.0e38f; int bi = N - 1;
    for (int i = 0; i < N; ++i) {
      float dx = P[3 * i + 0] - qx;
      float dy = P[3 * i + 1] - qy;
      float dz = P[3 * i + 2] - qz;
      float d  = dx * dx + dy * dy + dz * dz;
      bool beyond = (d > lastd) || (d == lastd && i > lasti);
      if (beyond && (d < bd || (d == bd && i < bi))) { bd = d; bi = i; }
    }
    out[(size_t)tid * KNN_K + j] = bi;
    lastd = bd; lasti = bi;
  }
}

// ---------------- fused SA layer: gather -> WMMA MLP -> relu -> max pool ------
// block = 128 threads = 4 waves, one sample point per wave.
// A staged in LDS in fragment order: [mt][kk] tiles of [lane][16] bf16.
template <int KPAD>
__global__ void point_conv_kernel(const float* __restrict__ src_xyz,
                                  const float* __restrict__ src_feat,
                                  const float* __restrict__ msg,
                                  const float* __restrict__ new_xyz,
                                  const int* __restrict__ knn,
                                  const __bf16* __restrict__ Wb,
                                  const float* __restrict__ bias,
                                  float* __restrict__ out_feat,
                                  int N, int S, int Cfeat, int use_feat0) {
  constexpr int KST = KPAD / 32;
  __shared__ __align__(32) __bf16 Wl[KST * 4 * 512];   // [kk][nt][lane][16]
  __shared__ __align__(32) __bf16 Al[4][KST * 2 * 512]; // per wave: [mt][kk][lane][16]
  int t = threadIdx.x, lane = t & 31, wv = t >> 5;

  __builtin_prefetch(Wb, 0, 1);
  for (int i = t; i < KST * 4 * 512; i += 128) Wl[i] = Wb[i];

  int gs = blockIdx.x * 4 + wv;        // global sample id in [0, B*S)
  int b  = gs / S;
  const float* q = new_xyz + (size_t)gs * 3;
  float qx = q[0], qy = q[1], qz = q[2];

  __bf16* A = Al[wv];
  {
    int row = lane;                    // this lane fills neighbor row `lane`
    int mt  = row >> 4;
    auto putA = [&](int K, float v) {
      int kk = K >> 5, kl = K & 31;
      A[((mt * KST + kk) << 9) + a_frag_off(row, kl)] = (__bf16)v;
    };
    int nb = knn[(size_t)gs * KNN_K + lane];
    const float* p = src_xyz + ((size_t)b * N + nb) * 3;
    float px = p[0], py = p[1], pz = p[2];
    putA(0, px - qx);
    putA(1, py - qy);
    putA(2, pz - qz);
    if (use_feat0) {                   // feat0 = concat(xyz, msg) on the fly
      putA(3, px); putA(4, py); putA(5, pz);
      for (int c = 3; c < Cfeat; ++c) putA(3 + c, msg[b * YDIM + (c - 3)]);
    } else {
      const float* f = src_feat + ((size_t)b * N + nb) * Cfeat;
      for (int c = 0; c < Cfeat; ++c) putA(3 + c, f[c]);
    }
    for (int c = 3 + Cfeat; c < KPAD; ++c) putA(c, 0.0f);
  }
  __syncthreads();

#pragma unroll
  for (int nt = 0; nt < 4; ++nt) {
    float cm = -3.0e38f;
#pragma unroll
    for (int mt = 0; mt < 2; ++mt) {   // 32 neighbor rows = 2 M-tiles
      v8f acc = {};
#pragma unroll
      for (int kk = 0; kk < KST; ++kk) {
        v16bf a = *reinterpret_cast<const v16bf*>(
            A + (((mt * KST + kk) << 9) + lane * 16));
        v16bf w = *reinterpret_cast<const v16bf*>(
            Wl + (((kk * 4 + nt) << 9) + lane * 16));
        acc = wmma_bf16(a, w, acc);
      }
#pragma unroll
      for (int r = 0; r < 8; ++r) cm = fmaxf(cm, acc[r]);
    }
    cm = fmaxf(cm, __shfl_xor(cm, 16, 32));   // combine M and M+8 lane halves
    if (lane < 16) {
      int n = nt * 16 + lane;
      out_feat[(size_t)gs * 64 + n] = fmaxf(cm + bias[n], 0.0f);
    }
  }
}

// ---------------- fused FP layer: 3-NN interp + concat + WMMA MLP -------------
// block = 128 threads = 4 waves; 64 points per block (16 rows per wave).
template <int KPAD, int NT>
__global__ void point_deconv_kernel(const float* __restrict__ xyz,
                                    const float* __restrict__ coarse_xyz,
                                    const float* __restrict__ skip_feat,
                                    const float* __restrict__ msg,
                                    const float* __restrict__ coarse_feat,
                                    const __bf16* __restrict__ Db,
                                    const float* __restrict__ bias,
                                    float* __restrict__ out,
                                    int Nl, int Sl, int Cskip,
                                    int Nout, int do_relu, int use_feat0) {
  constexpr int KST = KPAD / 32;
  __shared__ __align__(32) __bf16 Wl[KST * NT * 512]; // [kk][nt][lane][16]
  __shared__ __align__(32) __bf16 Rf[4][KST * 512];   // per wave: [kk][lane][16]
  int t = threadIdx.x, lane = t & 31, wv = t >> 5;

  __builtin_prefetch(Db, 0, 1);
  for (int i = t; i < KST * NT * 512; i += 128) Wl[i] = Db[i];

  int row0 = blockIdx.x * 64;           // global row = b*Nl + n
  if (t < 64) {
    int g = row0 + t;
    int b = g / Nl;
    const float* P = xyz + (size_t)g * 3;
    float px = P[0], py = P[1], pz = P[2];
    // 3-NN over Sl coarse points (stable ties like top_k)
    float d0 = 3.0e38f, d1 = 3.0e38f, d2 = 3.0e38f;
    int   i0 = 0, i1 = 0, i2 = 0;
    const float* C = coarse_xyz + (size_t)b * Sl * 3;
    for (int i = 0; i < Sl; ++i) {
      float dx = C[3 * i + 0] - px;
      float dy = C[3 * i + 1] - py;
      float dz = C[3 * i + 2] - pz;
      float d  = dx * dx + dy * dy + dz * dz;
      if (d < d0)      { d2 = d1; i2 = i1; d1 = d0; i1 = i0; d0 = d; i0 = i; }
      else if (d < d1) { d2 = d1; i2 = i1; d1 = d;  i1 = i; }
      else if (d < d2) { d2 = d;  i2 = i; }
    }
    float w0 = 1.0f / fmaxf(d0, 1e-10f);
    float w1 = 1.0f / fmaxf(d1, 1e-10f);
    float w2 = 1.0f / fmaxf(d2, 1e-10f);
    float ws = w0 + w1 + w2;
    w0 /= ws; w1 /= ws; w2 /= ws;

    __bf16* R = Rf[t >> 4];
    int r = t & 15;
    auto putR = [&](int K, float v) {
      int kk = K >> 5, kl = K & 31;
      R[(kk << 9) + a_frag_off(r, kl)] = (__bf16)v;
    };
    if (use_feat0) {                    // skip = concat(xyz, msg)
      putR(0, px); putR(1, py); putR(2, pz);
      for (int c = 3; c < Cskip; ++c) putR(c, msg[b * YDIM + (c - 3)]);
    } else {
      const float* f = skip_feat + (size_t)g * Cskip;
      for (int c = 0; c < Cskip; ++c) putR(c, f[c]);
    }
    const float* f0 = coarse_feat + ((size_t)b * Sl + i0) * 64;
    const float* f1 = coarse_feat + ((size_t)b * Sl + i1) * 64;
    const float* f2 = coarse_feat + ((size_t)b * Sl + i2) * 64;
    for (int c = 0; c < 64; ++c)
      putR(Cskip + c, w0 * f0[c] + w1 * f1[c] + w2 * f2[c]);
    for (int c = Cskip + 64; c < KPAD; ++c) putR(c, 0.0f);
  }
  __syncthreads();

  const __bf16* R = Rf[wv];
#pragma unroll
  for (int nt = 0; nt < NT; ++nt) {
    v8f acc = {};
#pragma unroll
    for (int kk = 0; kk < KST; ++kk) {
      v16bf a = *reinterpret_cast<const v16bf*>(R + ((kk << 9) + lane * 16));
      v16bf w = *reinterpret_cast<const v16bf*>(
          Wl + (((kk * NT + nt) << 9) + lane * 16));
      acc = wmma_bf16(a, w, acc);
    }
    int col = nt * 16 + (lane & 15);
    if (col < Nout) {
      float bb = bias[col];
#pragma unroll
      for (int r = 0; r < 8; ++r) {
        int m = r + ((lane & 16) ? 8 : 0);
        int g = row0 + wv * 16 + m;
        float v = acc[r] + bb;
        if (do_relu) v = fmaxf(v, 0.0f);
        out[(size_t)g * Nout + col] = v;
      }
    }
  }
}

// ---------------- host launcher ----------------
static inline size_t align256(size_t x) { return (x + 255) & ~(size_t)255; }

extern "C" void kernel_launch(void* const* d_in, const int* in_sizes, int n_in,
                              void* d_out, int out_size, void* d_ws, size_t ws_size,
                              hipStream_t stream) {
  const float* xyz = (const float*)d_in[0];
  const float* msg = (const float*)d_in[1];
  const float* W1  = (const float*)d_in[2];  const float* b1  = (const float*)d_in[3];
  const float* W2  = (const float*)d_in[4];  const float* b2  = (const float*)d_in[5];
  const float* W3  = (const float*)d_in[6];  const float* b3  = (const float*)d_in[7];
  const float* D1  = (const float*)d_in[8];  const float* db1 = (const float*)d_in[9];
  const float* D2  = (const float*)d_in[10]; const float* db2 = (const float*)d_in[11];
  const float* D3  = (const float*)d_in[12]; const float* db3 = (const float*)d_in[13];

  const int B = in_sizes[1] / YDIM;          // 64
  const int N = in_sizes[0] / (3 * B);       // 4096
  const int S1 = 256, S2 = 128, S3 = 64;
  const int CIN = 3 + YDIM;                  // 35

  // workspace bump allocator
  char* base = (char*)d_ws; size_t cur = 0;
  auto alloc = [&](size_t bytes) -> void* {
    void* p = base + cur; cur += align256(bytes); return p;
  };
  float* xyz1   = (float*)alloc((size_t)B * S1 * 3 * 4);
  float* xyz2   = (float*)alloc((size_t)B * S2 * 3 * 4);
  float* xyz3   = (float*)alloc((size_t)B * S3 * 3 * 4);
  float* feat1  = (float*)alloc((size_t)B * S1 * 64 * 4);
  float* feat2  = (float*)alloc((size_t)B * S2 * 64 * 4);
  float* feat3  = (float*)alloc((size_t)B * S3 * 64 * 4);
  float* featd2 = (float*)alloc((size_t)B * S2 * 64 * 4);
  float* featd1 = (float*)alloc((size_t)B * S1 * 64 * 4);
  int* fidx1 = (int*)alloc((size_t)B * S1 * 4);
  int* fidx2 = (int*)alloc((size_t)B * S2 * 4);
  int* fidx3 = (int*)alloc((size_t)B * S3 * 4);
  int* knn1  = (int*)alloc((size_t)B * S1 * KNN_K * 4);
  int* knn2  = (int*)alloc((size_t)B * S2 * KNN_K * 4);
  int* knn3  = (int*)alloc((size_t)B * S3 * KNN_K * 4);
  __bf16* W1b = (__bf16*)alloc((size_t)64  * 64 * 2);  // frag order, Kpad*Npad elems
  __bf16* W2b = (__bf16*)alloc((size_t)96  * 64 * 2);
  __bf16* W3b = (__bf16*)alloc((size_t)96  * 64 * 2);
  __bf16* D1b = (__bf16*)alloc((size_t)128 * 64 * 2);
  __bf16* D2b = (__bf16*)alloc((size_t)128 * 64 * 2);
  __bf16* D3b = (__bf16*)alloc((size_t)128 * 16 * 2);

  auto packs = [&](const float* W, __bf16* o, int K, int Nn, int NT, int Kpad) {
    int tot = (Kpad / 32) * NT * 512;
    pack_w_frag_kernel<<<(tot + 255) / 256, 256, 0, stream>>>(W, o, K, Nn, NT, tot);
  };
  packs(W1, W1b, CIN + 3, 64, 4, 64);        // K=38 -> Kpad 64
  packs(W2, W2b, 64 + 3, 64, 4, 96);         // K=67 -> Kpad 96
  packs(W3, W3b, 64 + 3, 64, 4, 96);
  packs(D1, D1b, 128, 64, 4, 128);
  packs(D2, D2b, 128, 64, 4, 128);
  packs(D3, D3b, CIN + 64, 3, 1, 128);       // K=99, Nout=3 -> one N-tile

  // ---- encoder level 1 ----
  fps_kernel<<<B, 256, 0, stream>>>(xyz, N, S1, fidx1, xyz1);
  knn_kernel<<<(B * S1 + 127) / 128, 128, 0, stream>>>(xyz1, xyz, S1, N, B * S1, knn1);
  point_conv_kernel<64><<<B * S1 / 4, 128, 0, stream>>>(
      xyz, nullptr, msg, xyz1, knn1, W1b, b1, feat1, N, S1, CIN, 1);
  // ---- encoder level 2 ----
  fps_kernel<<<B, 256, 0, stream>>>(xyz1, S1, S2, fidx2, xyz2);
  knn_kernel<<<(B * S2 + 127) / 128, 128, 0, stream>>>(xyz2, xyz1, S2, S1, B * S2, knn2);
  point_conv_kernel<96><<<B * S2 / 4, 128, 0, stream>>>(
      xyz1, feat1, msg, xyz2, knn2, W2b, b2, feat2, S1, S2, 64, 0);
  // ---- encoder level 3 ----
  fps_kernel<<<B, 256, 0, stream>>>(xyz2, S2, S3, fidx3, xyz3);
  knn_kernel<<<(B * S3 + 127) / 128, 128, 0, stream>>>(xyz3, xyz2, S3, S2, B * S3, knn3);
  point_conv_kernel<96><<<B * S3 / 4, 128, 0, stream>>>(
      xyz2, feat2, msg, xyz3, knn3, W3b, b3, feat3, S2, S3, 64, 0);

  // ---- decoder ----
  point_deconv_kernel<128, 4><<<B * S2 / 64, 128, 0, stream>>>(
      xyz2, xyz3, feat2, msg, feat3, D1b, db1, featd2, S2, S3, 64, 64, 1, 0);
  point_deconv_kernel<128, 4><<<B * S1 / 64, 128, 0, stream>>>(
      xyz1, xyz2, feat1, msg, featd2, D2b, db2, featd1, S1, S2, 64, 64, 1, 0);
  point_deconv_kernel<128, 1><<<B * N / 64, 128, 0, stream>>>(
      xyz, xyz1, nullptr, msg, featd1, D3b, db3, (float*)d_out,
      N, S1, CIN, 3, 0, 1);
  (void)n_in; (void)out_size; (void)ws_size;
}
